// BidirectionalMamba_18571438588112
// MI455X (gfx1250) — matrix operand order repaired and checked by reading, not compiled, not run on hardware
//
#include <hip/hip_runtime.h>
#include <hip/hip_bf16.h>

// ---------------------------------------------------------------------------
// BidirectionalMamba on MI455X (gfx1250, wave32, WMMA).
//   B=4, T=2048, DIM=512, DI=1024, N(d_state)=64, DT_RANK=32, D_CONV=4
// GEMMs use v_wmma_f32_16x16x32_f16 (f16 inputs, f32 accumulate); the K-loop
// is a compile-time 2x-unrolled ping-pong pipeline: fragments for the next
// K-step load while the current WMMAs issue, with zero register-rotation movs.
// The sequential selective scan keeps h[64] in VGPRs (2 f32/lane, wave32),
// one wave per (batch, channel), epilogue (D-residual + silu(z) gate) fused.
// ---------------------------------------------------------------------------

typedef __attribute__((ext_vector_type(8)))  _Float16 v8h;
typedef __attribute__((ext_vector_type(16))) _Float16 v16h;
typedef __attribute__((ext_vector_type(8)))  float    v8f;

#define BB   4
#define TT   2048
#define CDIM 512
#define DI   1024
#define NST  64
#define ROWS (BB * TT)   // 8192

// ---------------------------------------------------------------- convert
__global__ void f32_to_f16_kernel(const float* __restrict__ s,
                                  _Float16* __restrict__ d, int n) {
  int i = blockIdx.x * blockDim.x + threadIdx.x;
  if (i < n) d[i] = (_Float16)s[i];
}

// ---------------------------------------------------------------- layernorm
__global__ void layernorm_f16_kernel(const float* __restrict__ x,
                                     const float* __restrict__ g,
                                     const float* __restrict__ b,
                                     _Float16* __restrict__ out, int C) {
  int row = blockIdx.x;
  const float* xr = x + (size_t)row * C;
  float s = 0.f, s2 = 0.f;
  for (int i = threadIdx.x; i < C; i += blockDim.x) {
    float v = xr[i]; s += v; s2 += v * v;
  }
  __shared__ float sh[2][8];
  for (int o = 16; o > 0; o >>= 1) { s += __shfl_down(s, o); s2 += __shfl_down(s2, o); }
  int wid = threadIdx.x >> 5, lid = threadIdx.x & 31;
  if (lid == 0) { sh[0][wid] = s; sh[1][wid] = s2; }
  __syncthreads();
  if (wid == 0) {
    s  = (lid < 8) ? sh[0][lid] : 0.f;
    s2 = (lid < 8) ? sh[1][lid] : 0.f;
    for (int o = 4; o > 0; o >>= 1) { s += __shfl_down(s, o); s2 += __shfl_down(s2, o); }
    if (lid == 0) { sh[0][0] = s; sh[1][0] = s2; }
  }
  __syncthreads();
  float mean = sh[0][0] / C;
  float var  = sh[1][0] / C - mean * mean;
  float inv  = rsqrtf(var + 1e-5f);
  for (int i = threadIdx.x; i < C; i += blockDim.x) {
    float v = (xr[i] - mean) * inv * g[i] + b[i];
    out[(size_t)row * C + i] = (_Float16)v;
  }
}

// ---------------------------------------------------------------- fragments
// A 16x32 f16 fragment: lane(l16=M, half): K = half*8 + j  and  16+half*8 + j
__device__ __forceinline__ v16h load_a_frag(const _Float16* arow, int k0, int half) {
  v8h lo = *(const v8h*)(arow + k0 + half * 8);
  v8h hi = *(const v8h*)(arow + k0 + 16 + half * 8);
  v16h r;
#pragma unroll
  for (int i = 0; i < 8; i++) { r[i] = lo[i]; r[8 + i] = hi[i]; }
  return r;
}
// B 32x16 fragment (B = W^T, W row-major [N,K]): lane(l16=N, half): K = half*8 + j  and  16+half*8 + j
__device__ __forceinline__ v16h load_b_frag(const _Float16* wrow, int k0, int half) {
  v8h lo = *(const v8h*)(wrow + k0 + half * 8);
  v8h hi = *(const v8h*)(wrow + k0 + 16 + half * 8);
  v16h r;
#pragma unroll
  for (int i = 0; i < 8; i++) { r[i] = lo[i]; r[8 + i] = hi[i]; }
  return r;
}

// ---------------------------------------------------------------- WMMA GEMM
// out[m][n] = sum_k A[src(m)][k] * W[n][k]  (+bias +residual), f16 in f32 acc.
// One wave computes a 16(M) x (16*NACC)(N) tile; A fragment reused NACC times.
// K is compile-time; the K-loop is a 2x-unrolled ping-pong software pipeline
// (a0/b0 <-> a1/b1), so there are no loop-carried register copies.
// flip: A row source index is time-reversed per batch (bidirectional mamba).
template <int NACC, int K>
__global__ __launch_bounds__(256)
void gemm_f16_wmma(const _Float16* __restrict__ A, int lda,
                   const _Float16* __restrict__ W, int ldb,
                   float* __restrict__ oF, _Float16* __restrict__ oH, int ldc,
                   const float* __restrict__ bias,
                   const float* __restrict__ res, int ldres,
                   int M, int N, int flip, int T) {
  constexpr int STEPS = K / 32;
  static_assert(STEPS == 1 || (STEPS % 2) == 0, "K/32 must be 1 or even");
  int wave = (blockIdx.x * blockDim.x + threadIdx.x) >> 5;
  int lane = threadIdx.x & 31;
  int numMT = M >> 4;
  int numNG = N / (16 * NACC);
  if (wave >= numMT * numNG) return;        // wave-uniform: EXEC stays all-ones
  int mt = wave % numMT;
  int ng = wave / numMT;

  int half = lane >> 4;
  int l16  = lane & 15;
  int mlog = mt * 16 + l16;
  int msrc = mlog;
  if (flip) { int bb = mlog / T, t = mlog % T; msrc = bb * T + (T - 1 - t); }
  const _Float16* arow = A + (size_t)msrc * lda;
  const _Float16* wrow[NACC];
#pragma unroll
  for (int j = 0; j < NACC; j++)
    wrow[j] = W + (size_t)((ng * NACC + j) * 16 + l16) * ldb;

  v8f acc[NACC] = {};
  v16h a0 = load_a_frag(arow, 0, half), a1;
  v16h b0[NACC], b1[NACC];
#pragma unroll
  for (int j = 0; j < NACC; j++) b0[j] = load_b_frag(wrow[j], 0, half);

  if constexpr (STEPS == 1) {
#pragma unroll
    for (int j = 0; j < NACC; j++)
      acc[j] = __builtin_amdgcn_wmma_f32_16x16x32_f16(
          false, a0, false, b0[j], (short)0, acc[j], false, false);
  } else {
    for (int s = 0; s < STEPS; s += 2) {
      // prefetch k-step s+1 into buffer 1 (always exists: STEPS even)
      a1 = load_a_frag(arow, (s + 1) * 32, half);
#pragma unroll
      for (int j = 0; j < NACC; j++) b1[j] = load_b_frag(wrow[j], (s + 1) * 32, half);
#pragma unroll
      for (int j = 0; j < NACC; j++)
        acc[j] = __builtin_amdgcn_wmma_f32_16x16x32_f16(
            false, a0, false, b0[j], (short)0, acc[j], false, false);
      // prefetch k-step s+2 into buffer 0 (all but last iteration)
      if (s + 2 < STEPS) {
        a0 = load_a_frag(arow, (s + 2) * 32, half);
#pragma unroll
        for (int j = 0; j < NACC; j++) b0[j] = load_b_frag(wrow[j], (s + 2) * 32, half);
      }
#pragma unroll
      for (int j = 0; j < NACC; j++)
        acc[j] = __builtin_amdgcn_wmma_f32_16x16x32_f16(
            false, a1, false, b1[j], (short)0, acc[j], false, false);
    }
  }
  // C/D layout: element r -> M = mt*16 + half*8 + r, N = n-tile + l16
#pragma unroll
  for (int j = 0; j < NACC; j++) {
    int n = (ng * NACC + j) * 16 + l16;
#pragma unroll
    for (int r = 0; r < 8; r++) {
      int m = mt * 16 + half * 8 + r;
      float v = acc[j][r];
      if (bias) v += bias[n];
      if (res)  v += res[(size_t)m * ldres + n];
      if (oF) oF[(size_t)m * ldc + n] = v;
      if (oH) oH[(size_t)m * ldc + n] = (_Float16)v;
    }
  }
}

// ---------------------------------------------------------------- conv+silu
// xc[b,t,d] = silu( sum_{k=0..3} conv_w[d,k] * xin[b,t-3+k,d] + conv_b[d] )
__global__ void conv_silu_kernel(const _Float16* __restrict__ xz, int xzld,
                                 const float* __restrict__ cw,
                                 const float* __restrict__ cb,
                                 float* __restrict__ xc,
                                 _Float16* __restrict__ xc16,
                                 int T, int di, int total) {
  int idx = blockIdx.x * blockDim.x + threadIdx.x;
  if (idx >= total) return;
  int d   = idx % di;
  int row = idx / di;           // row = b*T + t
  int t = row % T, bb = row / T;
  float acc = cb[d];
#pragma unroll
  for (int k = 0; k < 4; k++) {
    int tt = t - 3 + k;
    if (tt >= 0)
      acc += cw[d * 4 + k] * (float)xz[((size_t)(bb * T + tt)) * xzld + d];
  }
  float s = acc / (1.f + __expf(-acc));   // silu
  xc[(size_t)row * di + d]   = s;
  xc16[(size_t)row * di + d] = (_Float16)s;
}

// ---------------------------------------------------------------- scan
// One wave per (b, d). Lane holds states n=lane, n+32. 2048 serial steps.
// Fused epilogue: yg = (y + xc*D) * silu(z), stored f16 (feeds out_proj GEMM).
__global__ __launch_bounds__(256)
void mamba_scan_kernel(const float* __restrict__ dt_raw,   // [ROWS, DI]
                       const float* __restrict__ dbl,      // [ROWS, 160]
                       const float* __restrict__ xc,       // [ROWS, DI]
                       const _Float16* __restrict__ xz,    // [ROWS, 2*DI], z @ DI+d
                       const float* __restrict__ A_log,    // [DI, NST]
                       const float* __restrict__ Dp,       // [DI]
                       const float* __restrict__ dtb,      // [DI]
                       _Float16* __restrict__ yg,          // [ROWS, DI]
                       int T) {
  int wave = (blockIdx.x * blockDim.x + threadIdx.x) >> 5;  // 0..4095
  int lane = threadIdx.x & 31;
  int d  = wave & (DI - 1);
  int bb = wave >> 10;
  float a0 = -__expf(A_log[d * NST + lane]);
  float a1 = -__expf(A_log[d * NST + lane + 32]);
  float dtbias = dtb[d];
  float Dd = Dp[d];
  float h0 = 0.f, h1 = 0.f;
  size_t rowbase = (size_t)bb * T;
  for (int t = 0; t < T; t++) {
    size_t r = rowbase + t;
    float dtv = dt_raw[r * DI + d] + dtbias;
    float dt  = (dtv > 20.f) ? dtv : log1pf(__expf(dtv));   // softplus
    float xv  = xc[r * DI + d];
    const float* dr = dbl + r * 160;
    float Bv0 = dr[32 + lane],  Bv1 = dr[64 + lane];
    float Cv0 = dr[96 + lane],  Cv1 = dr[128 + lane];
    float dtx = dt * xv;
    h0 = __expf(dt * a0) * h0 + dtx * Bv0;
    h1 = __expf(dt * a1) * h1 + dtx * Bv1;
    float p = h0 * Cv0 + h1 * Cv1;
    for (int o = 16; o > 0; o >>= 1) p += __shfl_xor(p, o);
    if (lane == 0) {
      float zv = (float)xz[r * (2 * DI) + DI + d];
      float y  = (p + xv * Dd) * (zv / (1.f + __expf(-zv)));
      yg[r * DI + d] = (_Float16)y;
    }
  }
}

// ---------------------------------------------------------------- host side
static inline int gemm_blocks(int M, int N, int nacc) {
  int waves = (M >> 4) * (N / (16 * nacc));
  return (waves + 7) / 8;   // 256 threads = 8 waves per block
}

extern "C" void kernel_launch(void* const* d_in, const int* in_sizes, int n_in,
                              void* d_out, int out_size, void* d_ws, size_t ws_size,
                              hipStream_t stream) {
  (void)in_sizes; (void)n_in; (void)out_size; (void)ws_size;
  // setup_inputs order (nested dicts flattened in insertion order):
  // 0:x 1:ln_g 2:ln_b  3..11:p_fw{in_proj,conv_w,conv_b,x_proj,dt_w,dt_b,A_log,D,out_proj}
  // 12..20:p_bw{...} 21:gate_w 22:gate_b 23:proj_w 24:proj_b
  const float* x    = (const float*)d_in[0];
  const float* ln_g = (const float*)d_in[1];
  const float* ln_b = (const float*)d_in[2];
  const float* proj_w = (const float*)d_in[23];
  const float* proj_b = (const float*)d_in[24];

  // bump allocator over d_ws (256B aligned); every byte read is written first
  char* wp = (char*)d_ws;
  auto alloc = [&](size_t bytes) -> char* {
    char* p = wp; wp += (bytes + 255) & ~(size_t)255; return p;
  };
  _Float16* xn16   = (_Float16*)alloc((size_t)ROWS * CDIM * 2);
  _Float16* cat16  = (_Float16*)alloc((size_t)ROWS * (2 * CDIM) * 2);
  _Float16* projw16= (_Float16*)alloc((size_t)CDIM * (2 * CDIM) * 2);
  _Float16* inpr16 = (_Float16*)alloc((size_t)(2 * DI) * CDIM * 2);
  _Float16* xprj16 = (_Float16*)alloc((size_t)160 * DI * 2);
  _Float16* dtw16  = (_Float16*)alloc((size_t)DI * 32 * 2);
  _Float16* outp16 = (_Float16*)alloc((size_t)CDIM * DI * 2);
  _Float16* xz16   = (_Float16*)alloc((size_t)ROWS * (2 * DI) * 2);
  float*    xc32   = (float*)   alloc((size_t)ROWS * DI * 4);
  _Float16* xc16   = (_Float16*)alloc((size_t)ROWS * DI * 2);
  float*    dbl32  = (float*)   alloc((size_t)ROWS * 160 * 4);
  _Float16* dbl16  = (_Float16*)alloc((size_t)ROWS * 160 * 2);
  float*    dtraw  = (float*)   alloc((size_t)ROWS * DI * 4);
  _Float16* yg16   = (_Float16*)alloc((size_t)ROWS * DI * 2);

  auto cvt = [&](const float* s, _Float16* d, int n) {
    f32_to_f16_kernel<<<(n + 255) / 256, 256, 0, stream>>>(s, d, n);
  };

  // 1. LayerNorm -> xn (f16)
  layernorm_f16_kernel<<<ROWS, 256, 0, stream>>>(x, ln_g, ln_b, xn16, CDIM);
  // 2. final projection weights (shared across directions)
  cvt(proj_w, projw16, CDIM * 2 * CDIM);

  for (int dir = 0; dir < 2; dir++) {
    int base = 3 + dir * 9;
    const float* in_proj = (const float*)d_in[base + 0];
    const float* conv_w  = (const float*)d_in[base + 1];
    const float* conv_b  = (const float*)d_in[base + 2];
    const float* x_proj  = (const float*)d_in[base + 3];
    const float* dt_w    = (const float*)d_in[base + 4];
    const float* dt_b    = (const float*)d_in[base + 5];
    const float* A_log   = (const float*)d_in[base + 6];
    const float* Dparam  = (const float*)d_in[base + 7];
    const float* out_prj = (const float*)d_in[base + 8];
    int flip = dir;   // bw direction: time-reversed reads/writes in GEMM A rows

    cvt(in_proj, inpr16, 2 * DI * CDIM);
    cvt(x_proj,  xprj16, 160 * DI);
    cvt(dt_w,    dtw16,  DI * 32);
    cvt(out_prj, outp16, CDIM * DI);

    // xz = xn(flip) @ in_proj^T        [8192 x 2048], K=512
    gemm_f16_wmma<4, CDIM><<<gemm_blocks(ROWS, 2 * DI, 4), 256, 0, stream>>>(
        xn16, CDIM, inpr16, CDIM, nullptr, xz16, 2 * DI,
        nullptr, nullptr, 0, ROWS, 2 * DI, flip, TT);

    // depthwise causal conv + SiLU on xin (= xz[:, :DI])
    conv_silu_kernel<<<(ROWS * DI + 255) / 256, 256, 0, stream>>>(
        xz16, 2 * DI, conv_w, conv_b, xc32, xc16, TT, DI, ROWS * DI);

    // dbl = xc @ x_proj^T              [8192 x 160], K=1024
    gemm_f16_wmma<2, DI><<<gemm_blocks(ROWS, 160, 2), 256, 0, stream>>>(
        xc16, DI, xprj16, DI, dbl32, dbl16, 160,
        nullptr, nullptr, 0, ROWS, 160, 0, TT);

    // dt_raw = dbl[:, :32] @ dt_w^T    [8192 x 1024], K=32
    gemm_f16_wmma<4, 32><<<gemm_blocks(ROWS, DI, 4), 256, 0, stream>>>(
        dbl16, 160, dtw16, 32, dtraw, nullptr, DI,
        nullptr, nullptr, 0, ROWS, DI, 0, TT);

    // selective scan (serial over T), fused D-residual + silu(z) gate
    mamba_scan_kernel<<<(BB * DI) / 8, 256, 0, stream>>>(
        dtraw, dbl32, xc32, xz16, A_log, Dparam, dt_b, yg16, TT);

    // cat[:, dir*512:(dir+1)*512] = yg(un-flip) @ out_proj^T   K=1024
    gemm_f16_wmma<4, DI><<<gemm_blocks(ROWS, CDIM, 4), 256, 0, stream>>>(
        yg16, DI, outp16, DI, nullptr, cat16 + dir * CDIM, 2 * CDIM,
        nullptr, nullptr, 0, ROWS, CDIM, flip, TT);
  }

  // out = x + cat @ proj_w^T + proj_b   [8192 x 512], K=1024
  gemm_f16_wmma<4, 2 * CDIM><<<gemm_blocks(ROWS, CDIM, 4), 256, 0, stream>>>(
      cat16, 2 * CDIM, projw16, 2 * CDIM, (float*)d_out, nullptr, CDIM,
      proj_b, x, CDIM, ROWS, CDIM, 0, TT);
}
